// MultiHeadAttention_32624571580450
// MI455X (gfx1250) — matrix operand order repaired and checked by reading, not compiled, not run on hardware
//
#include <hip/hip_runtime.h>

// ---------------------------------------------------------------------------
// MI455X (gfx1250) GQA attention block, bf16 WMMA everywhere.
//   B=1, T=2048, DIM=2048, 32 heads x 64 head_dim, 8 KV heads (GQA groups=4).
// Pipeline: cvt(f32->bf16) -> GEMM Q (pre-scaled by 0.125*log2e), K
//           -> GEMM V (transposed) -> flash attn (exp2 domain) -> GEMM out.
// Round 6: attention blocks = (query tile, 4 heads sharing one KV head);
//          K/V tiles staged once per block in double-buffered LDS via
//          global_load_async_to_lds_b128 -> 4x K/V traffic cut + overlap.
// ---------------------------------------------------------------------------

typedef __attribute__((ext_vector_type(16))) __bf16 v16bf;
typedef __attribute__((ext_vector_type(8)))  __bf16 v8bf;
typedef __attribute__((ext_vector_type(8)))  float  v8f;

constexpr int TS  = 2048;       // sequence length
constexpr int DM  = 2048;       // model dim
constexpr int NH  = 32;         // query heads
constexpr int HD  = 64;         // head dim
constexpr int KVH = 8;          // kv heads
constexpr int KVD = KVH * HD;   // 512

constexpr int LDS_STRIDE = 40;  // halves; 80B row stride (GEMM panels)
constexpr int KV_STRIDE  = 72;  // halves; 144B row stride (attn K/V panels)

// ---- fragment helpers -----------------------------------------------------

__device__ __forceinline__ v16bf ld16(const __bf16* p) {
  v16bf r;
  *(v8bf*)&r       = *(const v8bf*)p;
  *((v8bf*)&r + 1) = *(const v8bf*)(p + 8);
  return r;
}

// A matrix 16x32 bf16: lane holds row (lane&15); K {hi*16..+15}
__device__ __forceinline__ v16bf load_frag_a(const __bf16* base, int ld, int lane) {
  const int row = lane & 15, hi = lane >> 4;
  const __bf16* p = base + (size_t)row * ld + hi * 16;
  v16bf r;
  *(v8bf*)&r       = *(const v8bf*)p;
  *((v8bf*)&r + 1) = *(const v8bf*)(p + 8);
  return r;
}

// B matrix 32x16 bf16: lane holds column (lane&15); K rows {hi*16..+15}.
__device__ __forceinline__ v16bf load_frag_b(const __bf16* base, int ld, int lane) {
  const int col = lane & 15, hi = lane >> 4;
  return ld16(base + (size_t)col * ld + hi * 16);
}

__device__ __forceinline__ v8f wmma_bf16(v16bf a, v16bf b, v8f c) {
  return __builtin_amdgcn_wmma_f32_16x16x32_bf16(false, a, false, b, (short)0, c,
                                                 false, false);
}

// ---- async global->LDS copy (16B per lane), ASYNCcnt-tracked ---------------

__device__ __forceinline__ void async_cp16(const __bf16* g, __bf16* l) {
  const unsigned lds          = (unsigned)(unsigned long long)l;
  const unsigned long long ga = (unsigned long long)g;
  asm volatile("global_load_async_to_lds_b128 %0, %1, off"
               :: "v"(lds), "v"(ga) : "memory");
}

__device__ __forceinline__ void wait_async_all() {
  asm volatile("s_wait_asynccnt 0x0" ::: "memory");
}

// ---- single-instruction DPP16 xor butterflies (row-local, 16-lane domain) --

__device__ __forceinline__ float red_max16(float x) {
  float y;
  asm("v_max_num_f32_dpp %0, %1, %1 quad_perm:[1,0,3,2] row_mask:0xf bank_mask:0xf bound_ctrl:1"
      : "=v"(y) : "v"(x));
  asm("v_max_num_f32_dpp %0, %1, %1 quad_perm:[2,3,0,1] row_mask:0xf bank_mask:0xf bound_ctrl:1"
      : "=v"(x) : "v"(y));
  asm("v_max_num_f32_dpp %0, %1, %1 row_half_mirror row_mask:0xf bank_mask:0xf bound_ctrl:1"
      : "=v"(y) : "v"(x));
  asm("v_max_num_f32_dpp %0, %1, %1 row_mirror row_mask:0xf bank_mask:0xf bound_ctrl:1"
      : "=v"(x) : "v"(y));
  return x;
}

__device__ __forceinline__ float red_sum16(float x) {
  float y;
  asm("v_add_f32_dpp %0, %1, %1 quad_perm:[1,0,3,2] row_mask:0xf bank_mask:0xf bound_ctrl:1"
      : "=v"(y) : "v"(x));
  asm("v_add_f32_dpp %0, %1, %1 quad_perm:[2,3,0,1] row_mask:0xf bank_mask:0xf bound_ctrl:1"
      : "=v"(x) : "v"(y));
  asm("v_add_f32_dpp %0, %1, %1 row_half_mirror row_mask:0xf bank_mask:0xf bound_ctrl:1"
      : "=v"(y) : "v"(x));
  asm("v_add_f32_dpp %0, %1, %1 row_mirror row_mask:0xf bank_mask:0xf bound_ctrl:1"
      : "=v"(x) : "v"(y));
  return x;
}

// ---- f32 -> bf16 conversion ----------------------------------------------

__global__ __launch_bounds__(256) void cvt_bf16(const float* __restrict__ in,
                                                __bf16* __restrict__ out, int n) {
  const int i = (blockIdx.x * 256 + threadIdx.x) * 8;
  if (i >= n) return;
  const float4 f0 = *(const float4*)(in + i);
  const float4 f1 = *(const float4*)(in + i + 4);
  v8bf v;
  v[0] = (__bf16)f0.x; v[1] = (__bf16)f0.y; v[2] = (__bf16)f0.z; v[3] = (__bf16)f0.w;
  v[4] = (__bf16)f1.x; v[5] = (__bf16)f1.y; v[6] = (__bf16)f1.z; v[7] = (__bf16)f1.w;
  *(v8bf*)(out + i) = v;
}

// ---- GEMM: C[M,N] = oscale * (A[M,K] @ W[N,K]^T) ---------------------------
// 256 threads = 8 waves (4x2); block tile 128x128; wave tile 32x64.
// A/B staged as 128x32 bf16 LDS panels, double-buffered, async-filled.

template <int MODE>
__device__ __forceinline__ void store_tile(void* Cp, int M, int N, int m0, int n0,
                                           const v8f& c, float oscale, int lane) {
  const int ln = lane & 15, hi = lane >> 4;
  if constexpr (MODE == 0) {
    __bf16* C = (__bf16*)Cp;
#pragma unroll
    for (int r = 0; r < 8; ++r)
      C[(size_t)(m0 + hi * 8 + r) * N + n0 + ln] = (__bf16)(c[r] * oscale);
  } else if constexpr (MODE == 1) {
    __bf16* C = (__bf16*)Cp;
    v8bf v;
#pragma unroll
    for (int r = 0; r < 8; ++r) v[r] = (__bf16)(c[r] * oscale);
    *(v8bf*)(C + (size_t)(n0 + ln) * M + m0 + hi * 8) = v;  // 16B contiguous
  } else {
    float* C = (float*)Cp;
#pragma unroll
    for (int r = 0; r < 8; ++r)
      C[(size_t)(m0 + hi * 8 + r) * N + n0 + ln] = c[r] * oscale;
  }
}

template <int MODE>
__global__ __launch_bounds__(256) void gemm_xt(const __bf16* __restrict__ A,
                                               const __bf16* __restrict__ W,
                                               void* __restrict__ Cp,
                                               int M, int N, int K, float oscale) {
  const int tid  = threadIdx.x;
  const int lane = tid & 31;
  const int wv   = tid >> 5;
  const int wm   = wv & 3;
  const int wn   = wv >> 2;
  const int nbc  = N >> 7;
  const int mblk = (blockIdx.x / nbc) << 7;
  const int nblk = (blockIdx.x % nbc) << 7;

  __shared__ __bf16 sA[2][128][LDS_STRIDE];
  __shared__ __bf16 sB[2][128][LDS_STRIDE];

  const __bf16* Ag = A + (size_t)mblk * K;
  const __bf16* Bg = W + (size_t)nblk * K;

  auto stage = [&](int buf, int k0) {
#pragma unroll
    for (int i = 0; i < 2; ++i) {
      const int c = tid + i * 256;
      const int row = c >> 2, sub = (c & 3) * 8;
      async_cp16(Ag + (size_t)row * K + k0 + sub, &sA[buf][row][sub]);
    }
#pragma unroll
    for (int i = 0; i < 2; ++i) {
      const int c = tid + i * 256;
      const int row = c >> 2, sub = (c & 3) * 8;
      async_cp16(Bg + (size_t)row * K + k0 + sub, &sB[buf][row][sub]);
    }
  };

  v8f acc[2][4] = {};
  stage(0, 0);
  wait_async_all();
  __syncthreads();

  int cur = 0;
  for (int k0 = 0; k0 < K; k0 += 32) {
    if (k0 + 32 < K) stage(cur ^ 1, k0 + 32);

    v16bf a0 = load_frag_a(&sA[cur][wm * 32][0],      LDS_STRIDE, lane);
    v16bf a1 = load_frag_a(&sA[cur][wm * 32 + 16][0], LDS_STRIDE, lane);
    v16bf b0 = load_frag_b(&sB[cur][wn * 64][0],      LDS_STRIDE, lane);
    v16bf b1 = load_frag_b(&sB[cur][wn * 64 + 16][0], LDS_STRIDE, lane);
    v16bf b2 = load_frag_b(&sB[cur][wn * 64 + 32][0], LDS_STRIDE, lane);
    v16bf b3 = load_frag_b(&sB[cur][wn * 64 + 48][0], LDS_STRIDE, lane);

    acc[0][0] = wmma_bf16(a0, b0, acc[0][0]);
    acc[0][1] = wmma_bf16(a0, b1, acc[0][1]);
    acc[0][2] = wmma_bf16(a0, b2, acc[0][2]);
    acc[0][3] = wmma_bf16(a0, b3, acc[0][3]);
    acc[1][0] = wmma_bf16(a1, b0, acc[1][0]);
    acc[1][1] = wmma_bf16(a1, b1, acc[1][1]);
    acc[1][2] = wmma_bf16(a1, b2, acc[1][2]);
    acc[1][3] = wmma_bf16(a1, b3, acc[1][3]);

    wait_async_all();
    __syncthreads();
    cur ^= 1;
  }

#pragma unroll
  for (int i = 0; i < 2; ++i)
#pragma unroll
    for (int j = 0; j < 4; ++j)
      store_tile<MODE>(Cp, M, N, mblk + wm * 32 + i * 16,
                       nblk + wn * 64 + j * 16, acc[i][j], oscale, lane);
}

// ---- flash attention -------------------------------------------------------
// Block = (query tile qt, head group hg of 4 heads sharing kv head hg).
// 4 waves; wave w handles head hg*4+w. K/V tiles staged once per block in
// double-buffered LDS (async), shared by all 4 waves (identical trip counts).
// Q  : [T, DM]  bf16, PRE-SCALED by 0.125*log2(e)  (exp2 domain)
// Kc : [T, KVD] bf16 row-major ; Vt : [KVD, T] bf16 ; Ab : [T, DM] bf16

__global__ __launch_bounds__(128) void attn_kernel(const __bf16* __restrict__ Q,
                                                   const __bf16* __restrict__ Kc,
                                                   const __bf16* __restrict__ Vt,
                                                   __bf16* __restrict__ Ab) {
  const int tid  = threadIdx.x;
  const int lane = tid & 31;
  const int w    = tid >> 5;
  const int hg   = blockIdx.x & (KVH - 1);   // kv head (= head group)
  const int qt   = blockIdx.x >> 3;          // 0..127
  const int h    = hg * 4 + w;
  const int t0   = qt * 16;
  const int ln   = lane & 15;
  const int hi   = lane >> 4;

  // K tile: [s_local 64][d 64]; V tile: [d_local 64][s 64]; double-buffered.
  __shared__ __bf16 sK[2][64][KV_STRIDE];
  __shared__ __bf16 sV[2][64][KV_STRIDE];
  __shared__ __bf16 pb_all[4][16][KV_STRIDE];
  __bf16(*pb)[KV_STRIDE] = pb_all[w];

  const __bf16* kg = Kc + hg * HD;                 // + s*KVD + d
  const __bf16* vg = Vt + (size_t)hg * HD * TS;    // + d*TS + s

  // stage one 64-key step: 8KB K + 8KB V, 8x16B per thread
  auto stage = [&](int buf, int s0) {
#pragma unroll
    for (int i = 0; i < 4; ++i) {
      const int c = tid + i * 128;                 // 0..511
      const int row = c >> 3, sub = (c & 7) * 8;
      async_cp16(kg + (size_t)(s0 + row) * KVD + sub, &sK[buf][row][sub]);
    }
#pragma unroll
    for (int i = 0; i < 4; ++i) {
      const int c = tid + i * 128;
      const int row = c >> 3, sub = (c & 7) * 8;
      async_cp16(vg + (size_t)row * TS + s0 + sub, &sV[buf][row][sub]);
    }
  };

  const __bf16* qbase = Q + (size_t)t0 * DM + h * HD;
  const v16bf aq0 = load_frag_a(qbase,      DM, lane);  // d = 0..31
  const v16bf aq1 = load_frag_a(qbase + 32, DM, lane);  // d = 32..63

  v8f o[4] = {};
  float m_run[8], l_run[8];
#pragma unroll
  for (int r = 0; r < 8; ++r) { m_run[r] = -3.0e38f; l_run[r] = 0.0f; }

  // one 64-key block out of LDS buffer `buf`
  auto step64 = [&](int buf, int s0, bool masked) {
    // ---- S = Q K^T : four 16x16 col tiles ----
    v8f sa[4];
#pragma unroll
    for (int j = 0; j < 4; ++j) {
      const __bf16* kp = &sK[buf][16 * j + ln][0];
      v16bf b0 = ld16(kp + hi * 16);
      v16bf b1 = ld16(kp + 32 + hi * 16);
      v8f acc = {};
      acc = wmma_bf16(aq0, b0, acc);
      acc = wmma_bf16(aq1, b1, acc);
      sa[j] = acc;
    }

    // ---- online softmax in exp2 domain (row = t0 + hi*8 + r) ----
    float p[4][8];
#pragma unroll
    for (int r = 0; r < 8; ++r) {
      float v0 = sa[0][r], v1 = sa[1][r], v2 = sa[2][r], v3 = sa[3][r];
      if (masked) {
        const int trow = t0 + hi * 8 + r;
        if (s0 + ln      > trow) v0 = -3.0e38f;
        if (s0 + 16 + ln > trow) v1 = -3.0e38f;
        if (s0 + 32 + ln > trow) v2 = -3.0e38f;
        if (s0 + 48 + ln > trow) v3 = -3.0e38f;
      }
      const float mx   = red_max16(fmaxf(fmaxf(v0, v1), fmaxf(v2, v3)));
      const float mnew = fmaxf(m_run[r], mx);
      const float corr = __builtin_amdgcn_exp2f(m_run[r] - mnew);
      v0 = __builtin_amdgcn_exp2f(v0 - mnew);
      v1 = __builtin_amdgcn_exp2f(v1 - mnew);
      v2 = __builtin_amdgcn_exp2f(v2 - mnew);
      v3 = __builtin_amdgcn_exp2f(v3 - mnew);
      const float rs = red_sum16((v0 + v1) + (v2 + v3));
      l_run[r] = l_run[r] * corr + rs;
      m_run[r] = mnew;
      p[0][r] = v0; p[1][r] = v1; p[2][r] = v2; p[3][r] = v3;
      o[0][r] *= corr; o[1][r] *= corr; o[2][r] *= corr; o[3][r] *= corr;
    }

    // ---- transpose P (C layout) -> A layout via per-wave LDS ----
#pragma unroll
    for (int r = 0; r < 8; ++r) {
      __bf16* row = pb[hi * 8 + r];
      row[ln]      = (__bf16)p[0][r];
      row[16 + ln] = (__bf16)p[1][r];
      row[32 + ln] = (__bf16)p[2][r];
      row[48 + ln] = (__bf16)p[3][r];
    }
    asm volatile("s_wait_dscnt 0x0" ::: "memory");
    v16bf pa0, pa1;
    {
      const __bf16* pr = &pb[ln][0];
      *(v8bf*)&pa0       = *(const v8bf*)(pr + hi * 16);
      *((v8bf*)&pa0 + 1) = *(const v8bf*)(pr + 8 + hi * 16);
      *(v8bf*)&pa1       = *(const v8bf*)(pr + 32 + hi * 16);
      *((v8bf*)&pa1 + 1) = *(const v8bf*)(pr + 40 + hi * 16);
    }

    // ---- O += P @ V ----
#pragma unroll
    for (int c = 0; c < 4; ++c) {
      const __bf16* vp = &sV[buf][c * 16 + ln][0];
      v16bf bv0 = ld16(vp + hi * 16);         // keys s0    .. s0+31
      v16bf bv1 = ld16(vp + 32 + hi * 16);    // keys s0+32 .. s0+63
      o[c] = wmma_bf16(pa0, bv0, o[c]);
      o[c] = wmma_bf16(pa1, bv1, o[c]);
    }
  };

  // all 4 waves share t0 -> identical trip counts -> barriers are safe.
  // (t0>>6)*64 + 63 <= 2047, so staging loads stay in-bounds.
  const int nfull = t0 >> 6;
  stage(0, 0);
  wait_async_all();
  __syncthreads();

  int cur = 0;
  for (int ss = 0; ss < nfull; ++ss) {
    stage(cur ^ 1, (ss + 1) << 6);             // overlap next tile with math
    step64(cur, ss << 6, false);
    wait_async_all();
    __syncthreads();
    cur ^= 1;
  }
  step64(cur, nfull << 6, true);               // masked diagonal step

  // ---- normalize and store ----
#pragma unroll
  for (int c = 0; c < 4; ++c) {
#pragma unroll
    for (int r = 0; r < 8; ++r) {
      const float inv = 1.0f / l_run[r];
      Ab[(size_t)(t0 + hi * 8 + r) * DM + h * HD + c * 16 + ln] =
          (__bf16)(o[c][r] * inv);
    }
  }
}

// ---------------------------------------------------------------------------

extern "C" void kernel_launch(void* const* d_in, const int* in_sizes, int n_in,
                              void* d_out, int out_size, void* d_ws, size_t ws_size,
                              hipStream_t stream) {
  const float* x  = (const float*)d_in[0];
  const float* Wq = (const float*)d_in[1];
  const float* Wk = (const float*)d_in[2];
  const float* Wv = (const float*)d_in[3];
  const float* Wo = (const float*)d_in[4];
  // d_in[5] = causal mask (handled analytically)

  char* ws = (char*)d_ws;
  const size_t MB = 1ull << 20;
  __bf16* xb  = (__bf16*)(ws + 0 * MB);    // [T, DM]      8 MB
  __bf16* wqb = (__bf16*)(ws + 8 * MB);    // [DM, DM]     8 MB
  __bf16* wkb = (__bf16*)(ws + 16 * MB);   // [KVD, DM]    2 MB
  __bf16* wvb = (__bf16*)(ws + 18 * MB);   // [KVD, DM]    2 MB
  __bf16* wob = (__bf16*)(ws + 20 * MB);   // [DM, DM]     8 MB
  __bf16* Qb  = (__bf16*)(ws + 28 * MB);   // [T, DM]      8 MB
  __bf16* Kb  = (__bf16*)(ws + 36 * MB);   // [T, KVD]     2 MB
  __bf16* Vt  = (__bf16*)(ws + 38 * MB);   // [KVD, T]     2 MB
  __bf16* Ab  = (__bf16*)(ws + 40 * MB);   // [T, DM]      8 MB  (48 MB total)

  cvt_bf16<<<(TS * DM)  / (8 * 256), 256, 0, stream>>>(x,  xb,  TS * DM);
  cvt_bf16<<<(DM * DM)  / (8 * 256), 256, 0, stream>>>(Wq, wqb, DM * DM);
  cvt_bf16<<<(KVD * DM) / (8 * 256), 256, 0, stream>>>(Wk, wkb, KVD * DM);
  cvt_bf16<<<(KVD * DM) / (8 * 256), 256, 0, stream>>>(Wv, wvb, KVD * DM);
  cvt_bf16<<<(DM * DM)  / (8 * 256), 256, 0, stream>>>(Wo, wob, DM * DM);

  // softmax scale 1/sqrt(64) and log2(e) folded into Q projection (exp2 domain)
  const float qscale = 0.125f * 1.4426950408889634f;

  gemm_xt<0><<<(TS / 128) * (DM  / 128), 256, 0, stream>>>(xb, wqb, Qb, TS, DM,  DM, qscale);
  gemm_xt<0><<<(TS / 128) * (KVD / 128), 256, 0, stream>>>(xb, wkb, Kb, TS, KVD, DM, 1.0f);
  gemm_xt<1><<<(TS / 128) * (KVD / 128), 256, 0, stream>>>(xb, wvb, Vt, TS, KVD, DM, 1.0f);

  // attention: 128 query tiles x 8 kv-head groups = 1024 blocks, 128 threads
  attn_kernel<<<(TS / 16) * KVH, 128, 0, stream>>>(Qb, Kb, Vt, Ab);

  // out = Ab Wo^T  (f32 store to d_out)
  gemm_xt<2><<<(TS / 128) * (DM / 128), 256, 0, stream>>>(Ab, wob, d_out, TS, DM, DM, 1.0f);
}
